// GMN_51488067944931
// MI455X (gfx1250) — matrix operand order repaired and checked by reading, not compiled, not run on hardware
//
#include <hip/hip_runtime.h>
#include <math.h>

// ---------------------------------------------------------------------------
// CDNA5 / gfx1250 implementation of the GMN reference.
// Big GEMMs (cross-attention, GCN/update/gated matmuls) use
// v_wmma_f32_16x16x32_bf16 (bf16 in, f32 accumulate).  Attention is a
// flash-attention streaming softmax so the 8192x8192 score matrix is never
// materialized.  Weight matrices are pre-transposed to [N,K] so both A and B
// WMMA fragments load as contiguous b128s (no scalar gather/packing).
// ---------------------------------------------------------------------------

typedef __bf16 bf16;
typedef __attribute__((ext_vector_type(16))) __bf16 v16bf;
typedef __attribute__((ext_vector_type(8)))  __bf16 v8bf;
typedef __attribute__((ext_vector_type(8)))  float  v8f;

#define NN 8192
#define HH 256
#define DIN 128
#define EE 131072
#define TT 32
#define BB 16

__device__ __forceinline__ v16bf cat8(v8bf a, v8bf b) {
  return __builtin_shufflevector(a, b, 0,1,2,3,4,5,6,7,8,9,10,11,12,13,14,15);
}
__device__ __forceinline__ v8f wmma_bf16(v16bf a, v16bf b, v8f c) {
  return __builtin_amdgcn_wmma_f32_16x16x32_bf16(false, a, false, b, (short)0, c,
                                                 false, false);
}

// ------------------------------- elementwise -------------------------------
__global__ void k_f32_to_bf16(const float* __restrict__ s, bf16* __restrict__ d, int n) {
  int i = blockIdx.x * blockDim.x + threadIdx.x;
  if (i < n) d[i] = (bf16)s[i];
}
// W [K,N] f32  ->  Wt [N,K] bf16 (fused convert + transpose)
__global__ void k_cvt_transpose(const float* __restrict__ W, bf16* __restrict__ Wt,
                                int K, int N) {
  int idx = blockIdx.x * blockDim.x + threadIdx.x;
  if (idx < K * N) {
    int k = idx / N, n = idx % N;
    Wt[(size_t)n * K + k] = (bf16)W[idx];
  }
}
__global__ void k_copy_f32(const float* __restrict__ s, float* __restrict__ d, int n) {
  int i = blockIdx.x * blockDim.x + threadIdx.x;
  if (i < n) d[i] = s[i];
}
__global__ void k_fill_f32(float* p, float v, int n) {
  int i = blockIdx.x * blockDim.x + threadIdx.x;
  if (i < n) p[i] = v;
}
__global__ void k_rsqrt(float* p, int n) {
  int i = blockIdx.x * blockDim.x + threadIdx.x;
  if (i < n) p[i] = p[i] > 0.f ? rsqrtf(p[i]) : 0.f;
}

// --------------------------------- GCN -------------------------------------
__global__ void k_degree(const int* __restrict__ ei, float* __restrict__ deg) {
  int e = blockIdx.x * blockDim.x + threadIdx.x;
  if (e < EE) atomicAdd(&deg[ei[EE + e]], 1.0f);   // dst row of edge_index
}
// self-loop term: c = dinv^2 * h
__global__ void k_cinit(const float* __restrict__ h, const float* __restrict__ dinv,
                        float* __restrict__ c, int D) {
  int idx = blockIdx.x * blockDim.x + threadIdx.x;
  if (idx < NN * D) {
    int n = idx / D;
    c[idx] = dinv[n] * dinv[n] * h[idx];
  }
}
__global__ void k_scatter(const int* __restrict__ ei, const float* __restrict__ h,
                          const float* __restrict__ dinv, float* __restrict__ c, int D) {
  unsigned idx = blockIdx.x * blockDim.x + threadIdx.x;
  unsigned e = idx / D, j = idx % D;
  if (e < EE) {
    int s = ei[e], d = ei[EE + e];
    atomicAdd(&c[(size_t)d * D + j], dinv[s] * dinv[d] * h[(size_t)s * D + j]);
  }
}

// ---------------------- WMMA bf16 GEMM (C = A @ Bt^T) -----------------------
// A: [M,K] row-major bf16; Bt: [N,K] row-major bf16 (i.e. B transposed);
// C: [M,N] f32.  One full wave32 per 16x64 output tile: one A fragment feeds
// four WMMAs.  All fragment loads are contiguous 16B (b128) accesses.
__global__ __launch_bounds__(32) void k_gemm_bf16t(const bf16* __restrict__ A,
                                                   const bf16* __restrict__ Bt,
                                                   float* __restrict__ C,
                                                   int M, int N, int K) {
  int ntiles = N / 64;
  int tm = blockIdx.x / ntiles, tn = blockIdx.x % ntiles;
  int lane = threadIdx.x;
  int n = lane & 15, half = lane >> 4, kh = half * 8;
  v8f acc[4];
#pragma unroll
  for (int c = 0; c < 4; ++c)
#pragma unroll
    for (int r = 0; r < 8; ++r) acc[c][r] = 0.f;
  const bf16* Arow = A + (size_t)(tm * 16 + n) * K;
  const bf16* Brow0 = Bt + (size_t)(tn * 64 + n) * K;
  for (int k0 = 0; k0 < K; k0 += 32) {
    // A fragment: lane covers K = kh..kh+7 and 16+kh..16+kh+7
    v16bf af = cat8(*(const v8bf*)(Arow + k0 + kh),
                    *(const v8bf*)(Arow + k0 + 16 + kh));
#pragma unroll
    for (int c = 0; c < 4; ++c) {
      // B fragment: lane covers K = kh..kh+7 and 16+kh..16+kh+7 (same as A)
      const bf16* Br = Brow0 + (size_t)(c * 16) * K + k0 + half * 8;
      v16bf bf = cat8(*(const v8bf*)(Br), *(const v8bf*)(Br + 16));
      acc[c] = wmma_bf16(af, bf, acc[c]);
    }
  }
#pragma unroll
  for (int c = 0; c < 4; ++c) {
    float* Crow = C + (size_t)(tm * 16 + half * 8) * N + tn * 64 + c * 16 + n;
#pragma unroll
    for (int r = 0; r < 8; ++r) Crow[(size_t)r * N] = acc[c][r];
  }
}

__global__ void k_bias_act(const float* __restrict__ g, const float* __restrict__ b,
                           float* __restrict__ o, int D, int do_tanh) {
  int idx = blockIdx.x * blockDim.x + threadIdx.x;
  if (idx < NN * D) {
    float v = g[idx] + b[idx % D];
    o[idx] = do_tanh ? tanhf(v) : v;
  }
}

// X = [c | f | f-att] -> bf16, widths dc, df, df
__global__ void k_concat_bf16(const float* __restrict__ c, const float* __restrict__ f,
                              const float* __restrict__ att, bf16* __restrict__ X,
                              int dc, int df) {
  int Kt = dc + 2 * df;
  unsigned idx = blockIdx.x * blockDim.x + threadIdx.x;
  if (idx < (unsigned)NN * Kt) {
    unsigned nrow = idx / Kt, k = idx % Kt;
    float v;
    if (k < (unsigned)dc) v = c[(size_t)nrow * dc + k];
    else if (k < (unsigned)(dc + df)) v = f[(size_t)nrow * df + (k - dc)];
    else {
      unsigned q = k - dc - df;
      v = f[(size_t)nrow * df + q] - att[(size_t)nrow * df + q];
    }
    X[idx] = (bf16)v;
  }
}

// ----------------------- flash attention  O=softmax(QK^T)V ------------------
// Q,K,V: [N, D] bf16 row-major; O: [N, D] f32.  Block: 8 waves, each owns 16
// query rows (block covers 128); streams 32 keys/iteration through LDS.
template <int D>
__global__ __launch_bounds__(256) void k_fattn(const bf16* __restrict__ Q,
                                               const bf16* __restrict__ K,
                                               const bf16* __restrict__ V,
                                               float* __restrict__ O, int Nk) {
  constexpr int KB = 32;
  __shared__ bf16 kbuf[KB][D + 8];        // [key][feature]
  __shared__ bf16 vtb[D][KB + 8];         // [feature][key] (transposed V)
  __shared__ bf16 pbuf[8][16][KB + 8];    // per-wave P tile

  int tid = threadIdx.x;
  int wave = tid >> 5, lane = tid & 31;
  int n = lane & 15, half = lane >> 4, kh = half * 8;
  int qbase = blockIdx.x * 128 + wave * 16;

  // Q fragments, resident in VGPRs for the whole pass
  v16bf qf[D / 32];
  const bf16* Qrow = Q + (size_t)(qbase + n) * D;
#pragma unroll
  for (int dc = 0; dc < D / 32; ++dc) {
    v8bf a0 = *(const v8bf*)(Qrow + dc * 32 + kh);
    v8bf a1 = *(const v8bf*)(Qrow + dc * 32 + 16 + kh);
    qf[dc] = cat8(a0, a1);
  }

  v8f o[D / 16];
#pragma unroll
  for (int ct = 0; ct < D / 16; ++ct)
#pragma unroll
    for (int r = 0; r < 8; ++r) o[ct][r] = 0.f;
  float m[8], l[8];
#pragma unroll
  for (int r = 0; r < 8; ++r) { m[r] = -INFINITY; l[r] = 0.f; }

  for (int kb0 = 0; kb0 < Nk; kb0 += KB) {
    // cooperative stage of K (row-major) and V (transposed) into LDS
    constexpr int CH = (KB * D) / (256 * 8);
#pragma unroll
    for (int c = 0; c < CH; ++c) {
      int chunk = tid + c * 256;
      int kr = (chunk * 8) / D, col = (chunk * 8) % D;
      v8bf kd = *(const v8bf*)(K + (size_t)(kb0 + kr) * D + col);
      *(v8bf*)(&kbuf[kr][col]) = kd;
      v8bf vd = *(const v8bf*)(V + (size_t)(kb0 + kr) * D + col);
#pragma unroll
      for (int j = 0; j < 8; ++j) vtb[col + j][kr] = vd[j];
    }
    __syncthreads();
    if (kb0 + KB < Nk) {  // global_prefetch_b8 of the next block
      __builtin_prefetch(K + (size_t)(kb0 + KB) * D + tid * 8, 0, 0);
      __builtin_prefetch(V + (size_t)(kb0 + KB) * D + tid * 8, 0, 0);
    }

    // S = Q K^T for two 16-key tiles
    v8f s0, s1;
#pragma unroll
    for (int r = 0; r < 8; ++r) { s0[r] = 0.f; s1[r] = 0.f; }
#pragma unroll
    for (int dc = 0; dc < D / 32; ++dc) {
      v16bf b0 = cat8(*(const v8bf*)(&kbuf[n][dc * 32 + half * 8]),
                      *(const v8bf*)(&kbuf[n][dc * 32 + half * 8 + 16]));
      s0 = wmma_bf16(qf[dc], b0, s0);
      v16bf b1 = cat8(*(const v8bf*)(&kbuf[16 + n][dc * 32 + half * 8]),
                      *(const v8bf*)(&kbuf[16 + n][dc * 32 + half * 8 + 16]));
      s1 = wmma_bf16(qf[dc], b1, s1);
    }

    // online softmax: rows live across the 16 lanes of each half
    float sc[8];
#pragma unroll
    for (int r = 0; r < 8; ++r) {
      float t = fmaxf(s0[r], s1[r]);
      t = fmaxf(t, __shfl_xor(t, 1));
      t = fmaxf(t, __shfl_xor(t, 2));
      t = fmaxf(t, __shfl_xor(t, 4));
      t = fmaxf(t, __shfl_xor(t, 8));
      float mn = fmaxf(m[r], t);
      sc[r] = __expf(m[r] - mn);
      m[r] = mn;
    }
#pragma unroll
    for (int r = 0; r < 8; ++r) {
      float p0 = __expf(s0[r] - m[r]);
      float p1 = __expf(s1[r] - m[r]);
      float rs = p0 + p1;
      rs += __shfl_xor(rs, 1);
      rs += __shfl_xor(rs, 2);
      rs += __shfl_xor(rs, 4);
      rs += __shfl_xor(rs, 8);
      l[r] = l[r] * sc[r] + rs;
      pbuf[wave][half * 8 + r][n] = (bf16)p0;
      pbuf[wave][half * 8 + r][16 + n] = (bf16)p1;
    }
#pragma unroll
    for (int ct = 0; ct < D / 16; ++ct)
#pragma unroll
      for (int r = 0; r < 8; ++r) o[ct][r] *= sc[r];

    // same-wave LDS RAW: order the pbuf stores before the fragment reads
    asm volatile("s_wait_dscnt 0" ::: "memory");

    v16bf pf = cat8(*(const v8bf*)(&pbuf[wave][n][kh]),
                    *(const v8bf*)(&pbuf[wave][n][16 + kh]));
#pragma unroll
    for (int ct = 0; ct < D / 16; ++ct) {
      v16bf vb = cat8(*(const v8bf*)(&vtb[ct * 16 + n][half * 8]),
                      *(const v8bf*)(&vtb[ct * 16 + n][half * 8 + 16]));
      o[ct] = wmma_bf16(pf, vb, o[ct]);
    }
    __syncthreads();
  }

#pragma unroll
  for (int ct = 0; ct < D / 16; ++ct)
#pragma unroll
    for (int r = 0; r < 8; ++r)
      O[(size_t)(qbase + half * 8 + r) * D + ct * 16 + n] = o[ct][r] / l[r];
}

// ------------------------------- readout -----------------------------------
// p[j] = sum_n sigmoid(t[n,j]+gb[j]) * f[n,j];   one block per column.
__global__ void k_gated_pool(const float* __restrict__ t, const float* __restrict__ f,
                             const float* __restrict__ gb, float* __restrict__ p) {
  int j = blockIdx.x;
  float acc = 0.f;
  for (int nrow = threadIdx.x; nrow < NN; nrow += blockDim.x) {
    float s = 1.f / (1.f + __expf(-(t[(size_t)nrow * HH + j] + gb[j])));
    acc += s * f[(size_t)nrow * HH + j];
  }
  __shared__ float red[256];
  red[threadIdx.x] = acc;
  __syncthreads();
  for (int s = 128; s > 0; s >>= 1) {
    if (threadIdx.x < s) red[threadIdx.x] += red[threadIdx.x + s];
    __syncthreads();
  }
  if (threadIdx.x == 0) p[j] = red[0];
}

// ntn[t] = relu( p1^T W[:,:,t] p2 + Wb[t,:]@[p1;p2] + b[t] )
__global__ void k_ntn(const float* __restrict__ p1, const float* __restrict__ p2,
                      const float* __restrict__ W, const float* __restrict__ Wb,
                      const float* __restrict__ b, float* __restrict__ ntn) {
  int t = blockIdx.x;
  float acc = 0.f;
  for (int idx = threadIdx.x; idx < HH * HH; idx += blockDim.x) {
    int i = idx / HH, j = idx % HH;
    acc += p1[i] * W[(size_t)idx * TT + t] * p2[j];
  }
  __shared__ float red[256];
  red[threadIdx.x] = acc;
  __syncthreads();
  for (int s = 128; s > 0; s >>= 1) {
    if (threadIdx.x < s) red[threadIdx.x] += red[threadIdx.x + s];
    __syncthreads();
  }
  if (threadIdx.x == 0) {
    float blk = 0.f;
    for (int k = 0; k < 2 * HH; ++k)
      blk += Wb[(size_t)t * 2 * HH + k] * (k < HH ? p1[k] : p2[k - HH]);
    float v = red[0] + blk + b[t];
    ntn[t] = v > 0.f ? v : 0.f;
  }
}

__global__ void k_head(const float* __restrict__ ntn, const float* __restrict__ fc1W,
                       const float* __restrict__ fc1b, const float* __restrict__ scW,
                       const float* __restrict__ scb, const float* __restrict__ avgv,
                       float* __restrict__ out) {
  if (threadIdx.x == 0 && blockIdx.x == 0) {
    float acc = 0.f;
    for (int b = 0; b < BB; ++b) {
      float a = fc1b[b];
      for (int t = 0; t < TT; ++t) a += ntn[t] * fc1W[t * BB + b];
      acc += tanhf(a) * scW[b];
    }
    float score = 1.f / (1.f + __expf(-(acc + scb[0])));
    out[0] = score;
    out[1] = -logf(score) * avgv[0];
  }
}

// ------------------------------ orchestration ------------------------------
extern "C" void kernel_launch(void* const* d_in, const int* in_sizes, int n_in,
                              void* d_out, int out_size, void* d_ws, size_t ws_size,
                              hipStream_t stream) {
  (void)in_sizes; (void)n_in; (void)out_size; (void)ws_size;
  const float* feat1 = (const float*)d_in[0];
  const float* feat2 = (const float*)d_in[1];
  const int* ei1 = (const int*)d_in[2];
  const int* ei2 = (const int*)d_in[3];
  const float* avgv = (const float*)d_in[4];
  const float* gcnW[3] = {(const float*)d_in[5], (const float*)d_in[8], (const float*)d_in[11]};
  const float* updW[3] = {(const float*)d_in[6], (const float*)d_in[9], (const float*)d_in[12]};
  const float* updB[3] = {(const float*)d_in[7], (const float*)d_in[10], (const float*)d_in[13]};
  const float* gatedW = (const float*)d_in[14];
  const float* gatedB = (const float*)d_in[15];
  const float* tnW = (const float*)d_in[16];
  const float* tnWb = (const float*)d_in[17];
  const float* tnb = (const float*)d_in[18];
  const float* fc1W = (const float*)d_in[19];
  const float* fc1b = (const float*)d_in[20];
  const float* scW = (const float*)d_in[21];
  const float* scb = (const float*)d_in[22];
  float* out = (float*)d_out;

  char* ws = (char*)d_ws;
  size_t off = 0;
  auto alloc = [&](size_t bytes) -> void* {
    void* p = ws + off;
    off = (off + bytes + 255) & ~(size_t)255;
    return p;
  };
  float* f1 = (float*)alloc((size_t)NN * HH * 4);
  float* f2 = (float*)alloc((size_t)NN * HH * 4);
  bf16* f1b = (bf16*)alloc((size_t)NN * HH * 2);
  bf16* f2b = (bf16*)alloc((size_t)NN * HH * 2);
  float* h1 = (float*)alloc((size_t)NN * HH * 4);
  float* h2 = (float*)alloc((size_t)NN * HH * 4);
  float* c1 = (float*)alloc((size_t)NN * HH * 4);
  float* c2 = (float*)alloc((size_t)NN * HH * 4);
  float* at1 = (float*)alloc((size_t)NN * HH * 4);
  float* at2 = (float*)alloc((size_t)NN * HH * 4);
  float* g1 = (float*)alloc((size_t)NN * HH * 4);
  float* g2 = (float*)alloc((size_t)NN * HH * 4);
  bf16* X1 = (bf16*)alloc((size_t)NN * 768 * 2);
  bf16* X2 = (bf16*)alloc((size_t)NN * 768 * 2);
  float* dinv1 = (float*)alloc((size_t)NN * 4);
  float* dinv2 = (float*)alloc((size_t)NN * 4);
  bf16* Wgt = (bf16*)alloc((size_t)HH * HH * 2);   // [N=HH][K<=HH]
  bf16* Wut = (bf16*)alloc((size_t)HH * 768 * 2);  // [N=HH][K<=768]
  bf16* gWt = (bf16*)alloc((size_t)HH * HH * 2);
  float* tbuf = (float*)alloc((size_t)NN * HH * 4);
  float* p1 = (float*)alloc(HH * 4);
  float* p2 = (float*)alloc(HH * 4);
  float* ntnv = (float*)alloc(TT * 4);

  const int TPB = 256;
  auto blk = [](size_t n) { return (unsigned)((n + 255) / 256); };
  const int GEMM_GRID = (NN / 16) * (HH / 64);   // 16x64 tile per wave

  // degrees / symmetric norm (once; same every layer)
  k_fill_f32<<<blk(NN), TPB, 0, stream>>>(dinv1, 1.0f, NN);
  k_fill_f32<<<blk(NN), TPB, 0, stream>>>(dinv2, 1.0f, NN);
  k_degree<<<blk(EE), TPB, 0, stream>>>(ei1, dinv1);
  k_degree<<<blk(EE), TPB, 0, stream>>>(ei2, dinv2);
  k_rsqrt<<<blk(NN), TPB, 0, stream>>>(dinv1, NN);
  k_rsqrt<<<blk(NN), TPB, 0, stream>>>(dinv2, NN);

  // features into working buffers (never mutate inputs)
  k_copy_f32<<<blk((size_t)NN * DIN), TPB, 0, stream>>>(feat1, f1, NN * DIN);
  k_copy_f32<<<blk((size_t)NN * DIN), TPB, 0, stream>>>(feat2, f2, NN * DIN);

  int din[3] = {DIN, HH, HH};
  for (int L = 0; L < 3; ++L) {
    int di = din[L], Kt = 2 * di + HH;
    // bf16 copies: weights transposed to [N,K], features row-major
    k_cvt_transpose<<<blk((size_t)di * HH), TPB, 0, stream>>>(gcnW[L], Wgt, di, HH);
    k_f32_to_bf16<<<blk((size_t)NN * di), TPB, 0, stream>>>(f1, f1b, NN * di);
    k_f32_to_bf16<<<blk((size_t)NN * di), TPB, 0, stream>>>(f2, f2b, NN * di);
    // h = f @ Wg    (M=NN, N=HH, K=di)
    k_gemm_bf16t<<<GEMM_GRID, 32, 0, stream>>>(f1b, Wgt, h1, NN, HH, di);
    k_gemm_bf16t<<<GEMM_GRID, 32, 0, stream>>>(f2b, Wgt, h2, NN, HH, di);
    // GCN aggregation: self-loop + normalized edge scatter
    k_cinit<<<blk((size_t)NN * HH), TPB, 0, stream>>>(h1, dinv1, c1, HH);
    k_cinit<<<blk((size_t)NN * HH), TPB, 0, stream>>>(h2, dinv2, c2, HH);
    k_scatter<<<blk((size_t)EE * HH), TPB, 0, stream>>>(ei1, h1, dinv1, c1, HH);
    k_scatter<<<blk((size_t)EE * HH), TPB, 0, stream>>>(ei2, h2, dinv2, c2, HH);
    // cross attention: att1 = softmax_row(f1 f2^T) f2 ; att2 = softmax_row(f2 f1^T) f1
    if (di == 128) {
      k_fattn<128><<<NN / 128, 256, 0, stream>>>(f1b, f2b, f2b, at1, NN);
      k_fattn<128><<<NN / 128, 256, 0, stream>>>(f2b, f1b, f1b, at2, NN);
    } else {
      k_fattn<256><<<NN / 128, 256, 0, stream>>>(f1b, f2b, f2b, at1, NN);
      k_fattn<256><<<NN / 128, 256, 0, stream>>>(f2b, f1b, f1b, at2, NN);
    }
    // update: g = [c | f | f-att] @ Wu + bu  (tanh except last layer)
    k_cvt_transpose<<<blk((size_t)Kt * HH), TPB, 0, stream>>>(updW[L], Wut, Kt, HH);
    k_concat_bf16<<<blk((size_t)NN * Kt), TPB, 0, stream>>>(c1, f1, at1, X1, HH, di);
    k_concat_bf16<<<blk((size_t)NN * Kt), TPB, 0, stream>>>(c2, f2, at2, X2, HH, di);
    k_gemm_bf16t<<<GEMM_GRID, 32, 0, stream>>>(X1, Wut, g1, NN, HH, Kt);
    k_gemm_bf16t<<<GEMM_GRID, 32, 0, stream>>>(X2, Wut, g2, NN, HH, Kt);
    int act = (L < 2) ? 1 : 0;
    k_bias_act<<<blk((size_t)NN * HH), TPB, 0, stream>>>(g1, updB[L], f1, HH, act);
    k_bias_act<<<blk((size_t)NN * HH), TPB, 0, stream>>>(g2, updB[L], f2, HH, act);
  }

  // gated readout: p = sum_n sigmoid(f@W+b) * f
  k_cvt_transpose<<<blk((size_t)HH * HH), TPB, 0, stream>>>(gatedW, gWt, HH, HH);
  k_f32_to_bf16<<<blk((size_t)NN * HH), TPB, 0, stream>>>(f1, f1b, NN * HH);
  k_gemm_bf16t<<<GEMM_GRID, 32, 0, stream>>>(f1b, gWt, tbuf, NN, HH, HH);
  k_gated_pool<<<HH, 256, 0, stream>>>(tbuf, f1, gatedB, p1);
  k_f32_to_bf16<<<blk((size_t)NN * HH), TPB, 0, stream>>>(f2, f2b, NN * HH);
  k_gemm_bf16t<<<GEMM_GRID, 32, 0, stream>>>(f2b, gWt, tbuf, NN, HH, HH);
  k_gated_pool<<<HH, 256, 0, stream>>>(tbuf, f2, gatedB, p2);

  // NTN + scoring head
  k_ntn<<<TT, 256, 0, stream>>>(p1, p2, tnW, tnWb, tnb, ntnv);
  k_head<<<1, 32, 0, stream>>>(ntnv, fc1W, fc1b, scW, scb, avgv, out);
}
